// ExphormerAttention_43069932045063
// MI455X (gfx1250) — matrix operand order repaired and checked by reading, not compiled, not run on hardware
//
#include <hip/hip_runtime.h>
#include <hip/hip_bf16.h>

// ---------------------------------------------------------------------------
// Exphormer attention for MI455X (gfx1250, wave32).
//   Phase 1: QKV GEMM via v_wmma_f32_16x16x32_bf16 (bf16 hi/lo split ~= fp32)
//   Phase 2: per-edge score + exp + scatter (1 wave / edge, f32 atomics)
//   Phase 3: normalize by Z
// ---------------------------------------------------------------------------

#define N_NODES 50000
#define N_EDGES 800000
#define DIM 128
#define HEADS 8
#define HDIM 16

typedef __attribute__((ext_vector_type(16))) __bf16 v16bf;
typedef __attribute__((ext_vector_type(8)))  float  v8f;

// ---------------------------------------------------------------------------
// Phase 1: out = x @ W + b for W in {Wq,Wk,Wv}, one 16x16 tile per wave.
// Tiles: 3 matrices * (50000/16 = 3125) M-tiles * (128/16 = 8) N-tiles.
// bf16 split-precision: A=Ahi+Alo, B=Bhi+Blo; acc += Ahi*Bhi+Ahi*Blo+Alo*Bhi.
// ---------------------------------------------------------------------------
__global__ __launch_bounds__(256) void exph_qkv_gemm(
    const float* __restrict__ x,
    const float* __restrict__ Wq, const float* __restrict__ bq,
    const float* __restrict__ Wk, const float* __restrict__ bk,
    const float* __restrict__ Wv, const float* __restrict__ bv,
    float* __restrict__ Q, float* __restrict__ K, float* __restrict__ V)
{
    const int lane = threadIdx.x & 31;
    const int waveInBlk = threadIdx.x >> 5;
    const long tile = (long)blockIdx.x * 8 + waveInBlk;
    const long totalTiles = 3L * 3125L * 8L;
    if (tile >= totalTiles) return;

    const int mat = (int)(tile / (3125L * 8L));
    const int rem = (int)(tile % (3125L * 8L));
    const int mt  = rem >> 3;      // M tile (0..3124)
    const int nt  = rem & 7;       // N tile (0..7)

    const float* W; const float* bias; float* out;
    if (mat == 0)      { W = Wq; bias = bq; out = Q; }
    else if (mat == 1) { W = Wk; bias = bk; out = K; }
    else               { W = Wv; bias = bv; out = V; }

    const int half = lane >> 4;          // lane group (0 or 1)
    const int m16  = lane & 15;          // row within tile (A) / col (B,D)
    const int mrow = mt * 16 + m16;      // global row of x
    const int ncol = nt * 16 + m16;      // global col of W/out

    v8f acc = {};

    #pragma unroll
    for (int k0 = 0; k0 < DIM; k0 += 32) {
        // ---- A tile: x[mrow][k0 .. k0+31], ISA bf16 A-layout ----
        // element i of lane -> K = (i%8) + (i/8)*16 + half*8
        v16bf a_hi, a_lo;
        const float* xr = x + (long)mrow * DIM + k0 + half * 8;
        #pragma unroll
        for (int p = 0; p < 8; ++p) {           // contiguous float pairs
            const int kb = (p & 3) * 2 + (p >> 2) * 16; // 0,2,4,6,16,18,20,22
            const float f0 = xr[kb];
            const float f1 = xr[kb + 1];
            const __bf16 h0 = (__bf16)f0;
            const __bf16 h1 = (__bf16)f1;
            a_hi[2*p]   = h0;             a_hi[2*p+1] = h1;
            a_lo[2*p]   = (__bf16)(f0 - (float)h0);
            a_lo[2*p+1] = (__bf16)(f1 - (float)h1);
        }
        // ---- B tile: W[k][ncol], ISA bf16 B-layout ----
        // element i of lane -> K = k0 + (i%8) + (i/8)*16 + half*8, N = lane%16
        v16bf b_hi, b_lo;
        const float* wc = W + (long)(k0 + half * 8) * DIM + ncol;
        #pragma unroll
        for (int i = 0; i < 16; ++i) {
            const float f = wc[(long)((i & 7) + (i >> 3) * 16) * DIM];
            const __bf16 h = (__bf16)f;
            b_hi[i] = h;
            b_lo[i] = (__bf16)(f - (float)h);
        }
        // split-precision accumulate: hi*lo + lo*hi + hi*hi
        acc = __builtin_amdgcn_wmma_f32_16x16x32_bf16(false, a_hi, false, b_lo,
                                                      (short)0, acc, false, false);
        acc = __builtin_amdgcn_wmma_f32_16x16x32_bf16(false, a_lo, false, b_hi,
                                                      (short)0, acc, false, false);
        acc = __builtin_amdgcn_wmma_f32_16x16x32_bf16(false, a_hi, false, b_hi,
                                                      (short)0, acc, false, false);
    }

    // ---- D store: VGPR r, lane -> M = r + half*8, N = lane%16; add bias ----
    const float bv_ = bias[ncol];
    #pragma unroll
    for (int r = 0; r < 8; ++r) {
        const int row = mt * 16 + r + half * 8;
        out[(long)row * DIM + ncol] = acc[r] + bv_;
    }
}

// ---------------------------------------------------------------------------
// Phase 2: one wave per edge. Lane l covers channels [4l, 4l+4) (float4).
// Head h = lane>>2 (4 lanes * 4 ch = 16 ch = one head). Per-head dot via
// shfl_xor over the 4-lane group; exp(clamp(score/4)); f32 atomic scatter.
// E-feature computed on the fly: e[c] = attr * We[c] + be[c].
// ---------------------------------------------------------------------------
__global__ __launch_bounds__(256) void exph_edge_attn(
    const float* __restrict__ Q, const float* __restrict__ K,
    const float* __restrict__ V,
    const int*   __restrict__ src, const int* __restrict__ dst,
    const float* __restrict__ attr,
    const float* __restrict__ We, const float* __restrict__ be,
    float* __restrict__ wV, float* __restrict__ Z, int nEdges)
{
    const int lane = threadIdx.x & 31;
    const int edge = (int)(((long)blockIdx.x * blockDim.x + threadIdx.x) >> 5);
    if (edge >= nEdges) return;

    const int s = src[edge];
    const int d = dst[edge];
    const float a = attr[edge];
    const int c = lane * 4;

    const float4 kq = *(const float4*)(K + (long)s * DIM + c);
    const float4 qd = *(const float4*)(Q + (long)d * DIM + c);
    const float4 vv = *(const float4*)(V + (long)s * DIM + c);
    const float4 we = *(const float4*)(We + c);
    const float4 bb = *(const float4*)(be + c);

    const float e0 = fmaf(a, we.x, bb.x);
    const float e1 = fmaf(a, we.y, bb.y);
    const float e2 = fmaf(a, we.z, bb.z);
    const float e3 = fmaf(a, we.w, bb.w);

    float p = kq.x * qd.x * e0 + kq.y * qd.y * e1
            + kq.z * qd.z * e2 + kq.w * qd.w * e3;
    // reduce the 16-channel head dot across the 4-lane group
    p += __shfl_xor(p, 1, 32);
    p += __shfl_xor(p, 2, 32);

    float score = p * 0.25f;                       // / sqrt(16)
    score = fminf(fmaxf(score, -5.0f), 5.0f);
    const float sc = __expf(score);

    float* wvd = wV + (long)d * DIM + c;
    atomicAdd(wvd + 0, vv.x * sc);
    atomicAdd(wvd + 1, vv.y * sc);
    atomicAdd(wvd + 2, vv.z * sc);
    atomicAdd(wvd + 3, vv.w * sc);
    if ((lane & 3) == 0)
        atomicAdd(Z + (long)d * HEADS + (lane >> 2), sc);
}

// ---------------------------------------------------------------------------
// Phase 3: out[n][h*16+d] /= (Z[n][h] + 1e-6), in place on d_out.
// ---------------------------------------------------------------------------
__global__ __launch_bounds__(256) void exph_normalize(
    float* __restrict__ out, const float* __restrict__ Z, int total)
{
    const int i = blockIdx.x * blockDim.x + threadIdx.x;
    if (i >= total) return;
    const int node = i >> 7;            // /128
    const int h    = (i & 127) >> 4;    // /16
    out[i] = out[i] / (Z[node * HEADS + h] + 1e-6f);
}

// ---------------------------------------------------------------------------
extern "C" void kernel_launch(void* const* d_in, const int* in_sizes, int n_in,
                              void* d_out, int out_size, void* d_ws, size_t ws_size,
                              hipStream_t stream) {
    const float* x    = (const float*)d_in[0];
    const int*   eidx = (const int*)  d_in[1];   // [2, E]
    const float* attr = (const float*)d_in[2];   // [E, 1]
    // d_in[3..5]: virt_* unused (use_virt_nodes = False)
    const float* Wq = (const float*)d_in[6];
    const float* bq = (const float*)d_in[7];
    const float* Wk = (const float*)d_in[8];
    const float* bk = (const float*)d_in[9];
    const float* We = (const float*)d_in[10];
    const float* be = (const float*)d_in[11];
    const float* Wv = (const float*)d_in[12];
    const float* bv = (const float*)d_in[13];

    const int* src = eidx;
    const int* dst = eidx + N_EDGES;

    // workspace layout: Q | K | V | Z
    const size_t matElems = (size_t)N_NODES * DIM;
    float* Q = (float*)d_ws;
    float* K = Q + matElems;
    float* V = K + matElems;
    float* Z = V + matElems;
    float* wV = (float*)d_out;   // accumulate wV directly into d_out

    hipMemsetAsync(wV, 0, matElems * sizeof(float), stream);
    hipMemsetAsync(Z, 0, (size_t)N_NODES * HEADS * sizeof(float), stream);

    // Phase 1: 3 * 3125 * 8 = 75000 tiles, 8 waves (tiles) per 256-thread block
    exph_qkv_gemm<<<75000 / 8, 256, 0, stream>>>(x, Wq, bq, Wk, bk, Wv, bv, Q, K, V);

    // Phase 2: 1 wave per edge, 8 waves per block
    exph_edge_attn<<<(N_EDGES + 7) / 8, 256, 0, stream>>>(Q, K, V, src, dst, attr,
                                                          We, be, wV, Z, N_EDGES);

    // Phase 3
    const int total = (int)matElems;
    exph_normalize<<<(total + 255) / 256, 256, 0, stream>>>(wV, Z, total);
}
